// ScalarPotentialLMSymplectic_2319282340369
// MI455X (gfx1250) — matrix operand order repaired and checked by reading, not compiled, not run on hardware
//
#include <hip/hip_runtime.h>
#include <hip/hip_bf16.h>
#include <math.h>

typedef __bf16 bf16;
typedef __attribute__((ext_vector_type(16))) __bf16 v16bf;
typedef __attribute__((ext_vector_type(8)))  __bf16 v8bf;
typedef __attribute__((ext_vector_type(8)))  float  v8f;

#define BDIM 128
#define TILE 64
#define KTILE 32
#define PITCH 40   // bf16 elements; 80 bytes per row, 16B aligned

__device__ __forceinline__ float gelu_f(float x) {
  return 0.5f * x * (1.0f + erff(x * 0.7071067811865476f));
}
__device__ __forceinline__ float gelu_grad_f(float x) {
  return 0.5f * (1.0f + erff(x * 0.7071067811865476f))
       + x * 0.3989422804014327f * __expf(-0.5f * x * x);
}
__device__ __forceinline__ float softplus_f(float x) {
  return (x > 20.0f) ? x : log1pf(__expf(x));
}

// ---------------------------------------------------------------------------
// bf16 WMMA GEMM:  C[M,N] = A[M,K] * Bt[N,K]^T (+bias), fused epilogues.
// Both operands bf16, contiguous in K. Tiles staged with CDNA5 async
// copies (global_load_async_to_lds_b128, ASYNCcnt) into double-buffered LDS.
//   EPI 0 : a = acc+bias; outF = a (f32 pre-activation); outH = bf16(gelu(a))
//   EPI 1 : outH = bf16(acc * gelu'(aux))            (backward gelu mask)
//   EPI 2 : outF = acc + bias (f32 plain)
// Block: 128 thr = 4 waves; block tile 64x64; wave tile 32x32 (2x2 WMMA accs)
// M must be a multiple of 64; K a multiple of 32; N guarded (clamped tiles --
// clamped B rows only feed output columns >= N, which are never stored).
// ---------------------------------------------------------------------------
template<int EPI>
__global__ __launch_bounds__(BDIM)
void gemm_wmma(const bf16* __restrict__ Ap, const bf16* __restrict__ Bt,
               const float* __restrict__ bias, const float* __restrict__ aux,
               float* __restrict__ outF, bf16* __restrict__ outH,
               int M, int N, int K)
{
  __shared__ __align__(16) bf16 As[2][TILE][PITCH];   // [buf][m][k]
  __shared__ __align__(16) bf16 Bs[2][TILE][PITCH];   // [buf][n][k]

  const int tid  = threadIdx.x;
  const int lane = tid & 31;
  const int wave = tid >> 5;
  const int half = lane >> 4;   // lanes 16..31 -> 1
  const int l15  = lane & 15;

  const int m0 = blockIdx.y * TILE;
  const int n0 = blockIdx.x * TILE;
  const int wr = (wave >> 1) * 32;   // wave row offset inside block tile
  const int wc = (wave & 1) * 32;    // wave col offset inside block tile

  // staging map: 256 16-byte chunks per tile, 2 per thread
  const int r0  = tid >> 2;              // 0..31  (+32 for second chunk)
  const int c8  = (tid & 3) * 8;         // 0,8,16,24

  // issue one K-tile's async global->LDS transfers (4 x b128 per thread)
  auto issue_tile = [&](int kt, int buf) {
#pragma unroll
    for (int it = 0; it < 2; ++it) {
      const int r = r0 + it * 32;
      const bf16* ga = &Ap[(size_t)(m0 + r) * K + kt + c8];
      const unsigned la = (unsigned)(size_t)&As[buf][r][c8];
      asm volatile("global_load_async_to_lds_b128 %0, %1, off"
                   :: "v"(la), "v"(ga) : "memory");
      const int gn = (n0 + r < N) ? (n0 + r) : (N - 1);
      const bf16* gb = &Bt[(size_t)gn * K + kt + c8];
      const unsigned lb = (unsigned)(size_t)&Bs[buf][r][c8];
      asm volatile("global_load_async_to_lds_b128 %0, %1, off"
                   :: "v"(lb), "v"(gb) : "memory");
    }
  };

  v8f acc[2][2];
#pragma unroll
  for (int i = 0; i < 2; ++i)
#pragma unroll
    for (int j = 0; j < 2; ++j)
#pragma unroll
      for (int e = 0; e < 8; ++e) acc[i][j][e] = 0.0f;

  issue_tile(0, 0);

  int buf = 0;
  for (int k0 = 0; k0 < K; k0 += KTILE, buf ^= 1) {
    // own async transfers landed in LDS...
    asm volatile("s_wait_asynccnt 0x0" ::: "memory");
    // ...and everyone's (also: everyone done reading the buffer we reuse next)
    __syncthreads();
    if (k0 + KTILE < K) issue_tile(k0 + KTILE, buf ^ 1);

    // --- build fragments per CDNA5 16-bit WMMA layouts ---
    v16bf afr[2], bfr[2];
#pragma unroll
    for (int i = 0; i < 2; ++i) {
      const int row = wr + i * 16 + l15;
      const int ka  = half ? 8 : 0;              // lanes16-31: K=8..15 / 24..31
      v8bf lo = *(const v8bf*)&As[buf][row][ka];
      v8bf hi = *(const v8bf*)&As[buf][row][ka + 16];
#pragma unroll
      for (int e = 0; e < 8; ++e) { afr[i][e] = lo[e]; afr[i][e + 8] = hi[e]; }
    }
#pragma unroll
    for (int j = 0; j < 2; ++j) {
      const int col = wc + j * 16 + l15;
      const int kb  = half ? 8 : 0;              // lanes16-31: K=8..15 / 24..31
      v8bf lo = *(const v8bf*)&Bs[buf][col][kb];
      v8bf hi = *(const v8bf*)&Bs[buf][col][kb + 16];
#pragma unroll
      for (int e = 0; e < 8; ++e) { bfr[j][e] = lo[e]; bfr[j][e + 8] = hi[e]; }
    }
#pragma unroll
    for (int i = 0; i < 2; ++i)
#pragma unroll
      for (int j = 0; j < 2; ++j)
        acc[i][j] = __builtin_amdgcn_wmma_f32_16x16x32_bf16(
            false, afr[i], false, bfr[j], (short)0, acc[i][j], false, false);
  }

  // --- epilogue: C/D layout => row = wr + i*16 + half*8 + e, col = wc+j*16+l15
#pragma unroll
  for (int i = 0; i < 2; ++i) {
#pragma unroll
    for (int j = 0; j < 2; ++j) {
      const int gcol = n0 + wc + j * 16 + l15;
      if (gcol >= N) continue;
      const float bv = bias ? bias[gcol] : 0.0f;
#pragma unroll
      for (int e = 0; e < 8; ++e) {
        const int grow = m0 + wr + i * 16 + half * 8 + e;
        const size_t idx = (size_t)grow * N + gcol;
        const float aval = acc[i][j][e] + bv;
        if (EPI == 0)      { outF[idx] = aval; outH[idx] = (bf16)gelu_f(aval); }
        else if (EPI == 1) { outH[idx] = (bf16)(aval * gelu_grad_f(aux[idx])); }
        else               { outF[idx] = aval; }
      }
    }
  }
}

// ---------------------------------------------------------------------------
// prep: f32 -> bf16 copy
__global__ __launch_bounds__(256)
void splm_cvt_bf16(const float* __restrict__ src, bf16* __restrict__ dst, size_t n)
{
  const size_t i = (size_t)blockIdx.x * blockDim.x + threadIdx.x;
  if (i < n) dst[i] = (bf16)src[i];
}

// prep: Wt[n*K+k] = bf16(W[k*N+n])   (W is [K,N] row-major)
__global__ __launch_bounds__(256)
void splm_transpose_bf16(const float* __restrict__ W, bf16* __restrict__ Wt,
                         int K, int N)
{
  const size_t i = (size_t)blockIdx.x * blockDim.x + threadIdx.x;
  if (i >= (size_t)K * N) return;
  const int k = (int)(i / (size_t)N);
  const int n = (int)(i % (size_t)N);
  Wt[(size_t)n * K + k] = (bf16)W[i];
}

// h[b,t,:] = E[x[b,t],:] + P[t,:];  v = 0
__global__ __launch_bounds__(128)
void splm_embed(const int* __restrict__ x, const float* __restrict__ E,
                const float* __restrict__ P, float* __restrict__ h,
                float* __restrict__ v, int T, int D)
{
  const int bt = blockIdx.x;
  const int d  = threadIdx.x;
  const int t  = bt % T;
  const int tok = x[bt];
  const size_t idx = (size_t)bt * D + d;
  h[idx] = E[(size_t)tok * D + d] + P[(size_t)t * D + d];
  v[idx] = 0.0f;
}

// z[b,t,0:D] = bf16(cummean_t h); z[b,t,D:2D] = bf16(h)
__global__ __launch_bounds__(128)
void splm_cummean_concat(const float* __restrict__ h, bf16* __restrict__ z,
                         int T, int D)
{
  const int b = blockIdx.x;
  const int d = threadIdx.x;
  float run = 0.0f;
  for (int t = 0; t < T; ++t) {
    const float hv = h[((size_t)b * T + t) * D + d];
    run += hv;
    const size_t base = ((size_t)b * T + t) * (size_t)(2 * D);
    z[base + d]     = (bf16)(run / (float)(t + 1));
    z[base + D + d] = (bf16)hv;
  }
}

// g3[i,hc] = bf16( Wo[hc] * gelu'(a3[i,hc]) )
__global__ __launch_bounds__(256)
void splm_g3(const float* __restrict__ a3, const float* __restrict__ Wo,
             bf16* __restrict__ g3, size_t total, int H)
{
  const size_t i = (size_t)blockIdx.x * blockDim.x + threadIdx.x;
  if (i >= total) return;
  const int hc = (int)(i % (size_t)H);
  g3[i] = (bf16)(Wo[hc] * gelu_grad_f(a3[i]));
}

// f[b,t,d] = -( gz[b,t,D+d] + sum_{s>=t} gz[b,s,d]/(s+1) )   (cummean adjoint)
__global__ __launch_bounds__(128)
void splm_force_assemble(const float* __restrict__ gz, float* __restrict__ f,
                         int T, int D)
{
  const int b = blockIdx.x;
  const int d = threadIdx.x;
  float rs = 0.0f;
  for (int t = T - 1; t >= 0; --t) {
    const size_t base = ((size_t)b * T + t) * (size_t)(2 * D);
    rs += gz[base + d] / (float)(t + 1);
    f[((size_t)b * T + t) * D + d] = -(gz[base + D + d] + rs);
  }
}

// v += 0.5*f/m ; h += v     (DT = 1)
__global__ __launch_bounds__(256)
void splm_kick_drift(float* __restrict__ h, float* __restrict__ v,
                     const float* __restrict__ f, const float* __restrict__ raw_m,
                     size_t n)
{
  const size_t i = (size_t)blockIdx.x * blockDim.x + threadIdx.x;
  if (i >= n) return;
  const float m = softplus_f(raw_m[0]) + 0.001f;
  const float vv = v[i] + 0.5f * f[i] / m;
  v[i] = vv;
  h[i] += vv;
}

// v = (v + 0.5*f/m) * exp(-softplus(raw_gamma))
__global__ __launch_bounds__(256)
void splm_kick_damp(float* __restrict__ v, const float* __restrict__ f,
                    const float* __restrict__ raw_m, const float* __restrict__ raw_g,
                    size_t n)
{
  const size_t i = (size_t)blockIdx.x * blockDim.x + threadIdx.x;
  if (i >= n) return;
  const float m = softplus_f(raw_m[0]) + 0.001f;
  const float damp = __expf(-softplus_f(raw_g[0]));
  v[i] = (v[i] + 0.5f * f[i] / m) * damp;
}

// ---------------------------------------------------------------------------
extern "C" void kernel_launch(void* const* d_in, const int* in_sizes, int n_in,
                              void* d_out, int out_size, void* d_ws, size_t ws_size,
                              hipStream_t stream) {
  (void)in_sizes; (void)n_in; (void)out_size; (void)ws_size;
  const int*   x     = (const int*)  d_in[0];
  const float* E     = (const float*)d_in[1];
  const float* P     = (const float*)d_in[2];
  const float* W1    = (const float*)d_in[3];
  const float* b1    = (const float*)d_in[4];
  const float* W2    = (const float*)d_in[5];
  const float* b2    = (const float*)d_in[6];
  const float* W3    = (const float*)d_in[7];
  const float* b3    = (const float*)d_in[8];
  const float* Wo    = (const float*)d_in[9];
  // d_in[10] = bo: the gradient (force) and logits never depend on it.
  const float* raw_m = (const float*)d_in[11];
  const float* raw_g = (const float*)d_in[12];

  const int Bb = 8, T = 512, D = 128, H = 512, V = 50257;
  const int M = Bb * T;                 // 4096 tokens
  const size_t nhd = (size_t)M * D;     // h/v/f elements
  const size_t nmh = (size_t)M * H;     // activations

  uint8_t* wsp = (uint8_t*)d_ws;
  size_t off = 0;
  auto take = [&](size_t bytes) -> void* {
    void* p = wsp + off;
    off = (off + bytes + 255) & ~(size_t)255;
    return p;
  };
  float* h   = (float*)take(nhd * 4);
  float* v   = (float*)take(nhd * 4);
  float* f   = (float*)take(nhd * 4);
  bf16*  hb  = (bf16*) take(nhd * 2);                 // bf16 copy of h for logits
  bf16*  z   = (bf16*) take((size_t)M * 2 * D * 2);
  float* a1  = (float*)take(nmh * 4);
  float* a2  = (float*)take(nmh * 4);
  float* a3  = (float*)take(nmh * 4);
  bf16*  z1  = (bf16*) take(nmh * 2);
  bf16*  z2  = (bf16*) take(nmh * 2);
  bf16*  g3  = (bf16*) take(nmh * 2);
  bf16*  g2  = (bf16*) take(nmh * 2);
  bf16*  g1  = (bf16*) take(nmh * 2);
  float* gz  = (float*)take((size_t)M * 2 * D * 4);
  // bf16 weight copies: *t = transposed [N,K] for forward, *b = as-is for bwd
  bf16*  W1t = (bf16*) take((size_t)2 * D * H * 2);   // [H, 2D]
  bf16*  W2t = (bf16*) take((size_t)H * H * 2);       // [H, H]
  bf16*  W3t = (bf16*) take((size_t)H * H * 2);
  bf16*  W1b = (bf16*) take((size_t)2 * D * H * 2);   // [2D, H] row-major
  bf16*  W2b = (bf16*) take((size_t)H * H * 2);
  bf16*  W3b = (bf16*) take((size_t)H * H * 2);
  bf16*  Eb  = (bf16*) take((size_t)V * D * 2);       // [V, D]

  // ---- one-time prep (amortized over 9 force evals) ----
  {
    const size_t w1n = (size_t)2 * D * H, whn = (size_t)H * H, en = (size_t)V * D;
    splm_transpose_bf16<<<(unsigned)((w1n + 255) / 256), 256, 0, stream>>>(W1, W1t, 2 * D, H);
    splm_transpose_bf16<<<(unsigned)((whn + 255) / 256), 256, 0, stream>>>(W2, W2t, H, H);
    splm_transpose_bf16<<<(unsigned)((whn + 255) / 256), 256, 0, stream>>>(W3, W3t, H, H);
    splm_cvt_bf16<<<(unsigned)((w1n + 255) / 256), 256, 0, stream>>>(W1, W1b, w1n);
    splm_cvt_bf16<<<(unsigned)((whn + 255) / 256), 256, 0, stream>>>(W2, W2b, whn);
    splm_cvt_bf16<<<(unsigned)((whn + 255) / 256), 256, 0, stream>>>(W3, W3b, whn);
    splm_cvt_bf16<<<(unsigned)((en + 255) / 256), 256, 0, stream>>>(E, Eb, en);
  }

  const dim3 blk(BDIM);
  const dim3 gHH(H / TILE, M / TILE);                  // N=512
  const dim3 gHZ((2 * D) / TILE, M / TILE);            // N=256
  const dim3 gLG((V + TILE - 1) / TILE, M / TILE);     // logits

  auto force_eval = [&]() {
    splm_cummean_concat<<<Bb, D, 0, stream>>>(h, z, T, D);
    // forward MLP (store f32 pre-activations for the analytic backward)
    gemm_wmma<0><<<gHH, blk, 0, stream>>>(z,  W1t, b1, nullptr, a1, z1, M, H, 2 * D);
    gemm_wmma<0><<<gHH, blk, 0, stream>>>(z1, W2t, b2, nullptr, a2, z2, M, H, H);
    gemm_wmma<2><<<gHH, blk, 0, stream>>>(z2, W3t, b3, nullptr, a3, nullptr, M, H, H);
    // backward to inputs (Wo forward matvec never needed -- only its gradient)
    splm_g3<<<(unsigned)((nmh + 255) / 256), 256, 0, stream>>>(a3, Wo, g3, nmh, H);
    gemm_wmma<1><<<gHH, blk, 0, stream>>>(g3, W3b, nullptr, a2, nullptr, g2, M, H, H);
    gemm_wmma<1><<<gHH, blk, 0, stream>>>(g2, W2b, nullptr, a1, nullptr, g1, M, H, H);
    gemm_wmma<2><<<gHZ, blk, 0, stream>>>(g1, W1b, nullptr, nullptr, gz, nullptr, M, 2 * D, H);
    splm_force_assemble<<<Bb, D, 0, stream>>>(gz, f, T, D);
  };

  // h = E[x] + P ; v = 0 ; f = force(h)
  splm_embed<<<M, D, 0, stream>>>(x, E, P, h, v, T, D);
  force_eval();

  // 8 leapfrog steps (DT=1; uniform full damping on v as noted in reference)
  const unsigned gN = (unsigned)((nhd + 255) / 256);
  for (int s = 0; s < 8; ++s) {
    splm_kick_drift<<<gN, 256, 0, stream>>>(h, v, f, raw_m, nhd);
    force_eval();
    splm_kick_damp<<<gN, 256, 0, stream>>>(v, f, raw_m, raw_g, nhd);
  }

  // logits = h @ E^T   (Eb is [V,D] row-major = contiguous-in-K "Bt" operand)
  splm_cvt_bf16<<<gN, 256, 0, stream>>>(h, hb, nhd);
  gemm_wmma<2><<<gLG, blk, 0, stream>>>(hb, Eb, nullptr, nullptr,
                                        (float*)d_out, nullptr, M, V, D);
}
